// Attention_model_25967372271699
// MI455X (gfx1250) — matrix operand order repaired and checked by reading, not compiled, not run on hardware
//
#include <hip/hip_runtime.h>
#include <hip/hip_bf16.h>

#define Bc 64
#define Lc 1024
#define Dc 1024
#define Uc 1024
#define ROWS_TOTAL (Bc * Lc)        // 65536
#define MTILE 128                   // rows per workgroup (8 waves x 16-row strips)
#define NTHREADS 256

// LDS fragment-order layout: [j][q][lane][8 bf16] with padded strides so ds reads
// are contiguous 512B per op (conflict-free) and use immediate offsets.
#define STRIDE_Q 264                               // elems (16B slot * 32 lanes + 8 pad)
#define STRIDE_J (2 * STRIDE_Q)                    // 528 elems = 1056 B per k-step
#define STRIDE_STRIP (32 * STRIDE_J)               // 16896 elems per 16-row strip
#define SA_ELEMS (8 * STRIDE_STRIP)                // 135168 elems (8 strips)
#define HALF_J 16                                  // k-steps per pipeline phase
#define HALF_ELEMS (HALF_J * STRIDE_J)             // 8448 elems per B half-buffer
#define SB_ELEMS (2 * HALF_ELEMS)                  // 16896 elems (double-buffered halves)
#define SPV_OFF (SA_ELEMS + SB_ELEMS)              // bf16-elem offset of float2 ph/V table
#define LDS_BYTES ((SA_ELEMS + SB_ELEMS) * 2 + Uc * 2 * 4)   // 312320 B <= 320 KB

typedef __attribute__((ext_vector_type(16))) __bf16 v16bf;
typedef __attribute__((ext_vector_type(8)))  float  v8f;
typedef __attribute__((ext_vector_type(4)))  int    v4i;

union ABFrag { uint4 q[2]; v16bf v; };

#if defined(__has_builtin)
# if __has_builtin(__builtin_amdgcn_global_load_async_to_lds_b128)
#  define HAVE_ASYNC_LDS 1
# endif
#endif

__device__ __forceinline__ float fast_tanhf(float x) {
#if defined(__has_builtin) && __has_builtin(__builtin_amdgcn_tanhf)
    return __builtin_amdgcn_tanhf(x);            // V_TANH_F32 (CDNA5 trans op)
#elif defined(__has_builtin) && __has_builtin(__builtin_amdgcn_tanh_f32)
    return __builtin_amdgcn_tanh_f32(x);
#else
    return tanhf(x);
#endif
}

__device__ __forceinline__ unsigned short f2bf(float f) {
    // round-to-nearest-even f32 -> bf16
    unsigned u = __float_as_uint(f);
    unsigned r = ((u >> 16) & 1u) + 0x7fffu;
    return (unsigned short)((u + r) >> 16);
}

// 16 k-steps (K=512) of WMMA from fragment-ordered LDS, immediate offsets only.
__device__ __forceinline__ v8f half_mma(const unsigned short* __restrict__ aP,
                                        const unsigned short* __restrict__ bP,
                                        v8f acc) {
    #pragma unroll 4
    for (int j = 0; j < HALF_J; ++j) {
        ABFrag a, bb;
        a.q[0]  = *(const uint4*)(aP + j * STRIDE_J);
        a.q[1]  = *(const uint4*)(aP + j * STRIDE_J + STRIDE_Q);
        bb.q[0] = *(const uint4*)(bP + j * STRIDE_J);
        bb.q[1] = *(const uint4*)(bP + j * STRIDE_J + STRIDE_Q);
        acc = __builtin_amdgcn_wmma_f32_16x16x32_bf16(
                  false, a.v, false, bb.v, (short)0, acc, false, false);
    }
    return acc;
}

// ---------------- Kernel 1: W1 [D][U] f32  ->  W1T [U][D] bf16 ----------------
__global__ void __launch_bounds__(256) transpose_w1_kernel(const float* __restrict__ W1,
                                                           unsigned short* __restrict__ W1T) {
    __shared__ float tile[32][33];
    const int tx = threadIdx.x;           // 0..31
    const int ty = threadIdx.y;           // 0..7
    const int u0 = blockIdx.x * 32;
    const int d0 = blockIdx.y * 32;
    #pragma unroll
    for (int i = 0; i < 32; i += 8)
        tile[ty + i][tx] = W1[(size_t)(d0 + ty + i) * Uc + (u0 + tx)];
    __syncthreads();
    #pragma unroll
    for (int i = 0; i < 32; i += 8)
        W1T[(size_t)(u0 + ty + i) * Dc + (d0 + tx)] = f2bf(tile[tx][ty + i]);
}

// ------------- Kernel 2: ph[b][u] = hidden[b]·W2[:,u] + b2[u] + b1[u] -------------
__global__ void __launch_bounds__(256) proj_h_kernel(const float* __restrict__ hidden,
                                                     const float* __restrict__ W2,
                                                     const float* __restrict__ b2,
                                                     const float* __restrict__ b1,
                                                     float* __restrict__ ph) {
    const int u = blockIdx.x * 256 + threadIdx.x;
    const int b = blockIdx.y;
    const float* h = hidden + (size_t)b * Dc;
    float acc = 0.f;
    #pragma unroll 4
    for (int k = 0; k < Dc; ++k)
        acc += h[k] * W2[(size_t)k * Uc + u];
    ph[(size_t)b * Uc + u] = acc + b2[u] + b1[u];
}

// ------- Kernel 3: fused logits[b,l] = tanh(F@W1 + ph)·V + bV via bf16 WMMA -------
__global__ void __launch_bounds__(NTHREADS) attn_logits_wmma(const float* __restrict__ features,
                                                             const unsigned short* __restrict__ W1T,
                                                             const float* __restrict__ ph,
                                                             const float* __restrict__ V,
                                                             const float* __restrict__ bV,
                                                             float* __restrict__ logits) {
    extern __shared__ unsigned short smem[];
    unsigned short* sA  = smem;              // 8 strips of fragment-ordered A
    unsigned short* sB  = smem + SA_ELEMS;   // two fragment-ordered B half-buffers
    float*          sPV = (float*)(smem + SPV_OFF);  // interleaved {ph+biases, V} per u

    const int tid   = threadIdx.x;
    const int wave  = tid >> 5;       // 0..7  -> M strip of 16 rows
    const int lane  = tid & 31;
    const int half  = lane >> 4;      // 0 or 1
    const int col   = lane & 15;      // N within 16-wide tile
    const int mbase = blockIdx.x * MTILE;
    const int b     = mbase >> 10;    // 128 | 1024 -> whole tile in one batch

    // Stage half a W1T ntile-slab (16 u-rows x 512 K) into buffer hb via async DMA.
    // B fragment convention: rem = 2q + h so q0 holds K=32j+8h.., q1 holds +16.
    // Fixed trip count (4 issues/thread), fully unrolled: straight-line DMA issue.
    auto stage_b = [&](int nt, int hb) {
        const unsigned short* wsrc = W1T + (size_t)(nt * 16) * Dc + hb * (HALF_J * 32);
        unsigned short* dbuf = sB + hb * HALF_ELEMS;
        #pragma unroll
        for (int i = 0; i < (16 * 64) / NTHREADS; ++i) {
            const int c   = i * NTHREADS + tid;
            const int k8  = c & 63;             // 16B chunk within half (K=512)
            const int row = c >> 6;
            const int j = k8 >> 2, rem = k8 & 3;
            const int h = rem & 1, q = rem >> 1;
            const int dst = j * STRIDE_J + q * STRIDE_Q + (((h << 4) | row) << 3);
#ifdef HAVE_ASYNC_LDS
            __builtin_amdgcn_global_load_async_to_lds_b128(
                (v4i*)(wsrc + (size_t)row * Dc + k8 * 8),
                (v4i*)(dbuf + dst), 0, 0);
#else
            *(uint4*)(dbuf + dst) = *(const uint4*)(wsrc + (size_t)row * Dc + k8 * 8);
#endif
        }
    };

    stage_b(0, 0);   // first phase's DMA overlaps the feature-convert loop below

    // ---- stage features tile (f32 -> bf16) into fragment-ordered LDS ----
    // Chunk (row,k8) holds K=[8*k8,+8). A fragment: rem=k8&3 -> h=rem&1, q=rem>>1
    // so q0 holds K=32j+8h.., q1 holds K=32j+16+8h.. (ISA 16-bit A table).
    for (int i = 0; i < (MTILE * (Dc / 8)) / NTHREADS; ++i) {
        const int c   = i * NTHREADS + tid;
        const int k8  = c & 127;
        const int row = c >> 7;
        const float4* src = (const float4*)(features + (size_t)(mbase + row) * Dc);
        const float4 f0 = src[k8 * 2];
        const float4 f1 = src[k8 * 2 + 1];
        uint4 pk;
        pk.x = (unsigned)f2bf(f0.x) | ((unsigned)f2bf(f0.y) << 16);
        pk.y = (unsigned)f2bf(f0.z) | ((unsigned)f2bf(f0.w) << 16);
        pk.z = (unsigned)f2bf(f1.x) | ((unsigned)f2bf(f1.y) << 16);
        pk.w = (unsigned)f2bf(f1.z) | ((unsigned)f2bf(f1.w) << 16);
        const int strip = row >> 4, m = row & 15;
        const int j = k8 >> 2, rem = k8 & 3;
        const int h = rem & 1, q = rem >> 1;
        *(uint4*)(sA + strip * STRIDE_STRIP + j * STRIDE_J + q * STRIDE_Q
                     + (((h << 4) | m) << 3)) = pk;
    }
    // ---- stage {ph, V} pairs so the epilogue is a single ds_load_b64 ----
    #pragma unroll
    for (int i = 0; i < Uc / NTHREADS; ++i) {
        const int u = i * NTHREADS + tid;
        sPV[2 * u]     = ph[(size_t)b * Uc + u];
        sPV[2 * u + 1] = V[u];
    }

#ifdef HAVE_ASYNC_LDS
    asm volatile("s_wait_asynccnt 0" ::: "memory");
#endif
    __syncthreads();

    const unsigned short* aPtr = sA + wave * STRIDE_STRIP + lane * 8;
    const unsigned short* bPtr = sB + lane * 8;

    float partial[8];
    #pragma unroll
    for (int r = 0; r < 8; ++r) partial[r] = 0.f;
    const float bVv = bV[0];

    // Software pipeline, half-ntile phases: issue next DMA, compute current half,
    // then s_wait_asynccnt 0 + barrier (DMA had a full compute-half to land).
    for (int nt = 0; nt < Uc / 16; ++nt) {
        v8f acc = {};
        stage_b(nt, 1);                                    // DMA hi half of this ntile
        acc = half_mma(aPtr, bPtr, acc);                   // K 0..511 from buf0
#ifdef HAVE_ASYNC_LDS
        asm volatile("s_wait_asynccnt 0" ::: "memory");
#endif
        __syncthreads();
        if (nt + 1 < Uc / 16) stage_b(nt + 1, 0);          // DMA lo half of next ntile
        acc = half_mma(aPtr + HALF_J * STRIDE_J, bPtr + HALF_ELEMS, acc);  // K 512..1023

        // ---- epilogue: tanh + dot with V, accumulate per-row logit partials ----
        const int u = nt * 16 + col;
        const float2 tv = *(const float2*)(sPV + 2 * u);   // {ph+b1+b2, V[u]}
        #pragma unroll
        for (int r = 0; r < 8; ++r)
            partial[r] += fast_tanhf(acc[r] + tv.x) * tv.y;

#ifdef HAVE_ASYNC_LDS
        asm volatile("s_wait_asynccnt 0" ::: "memory");
#endif
        __syncthreads();
    }

    // reduce across the 16 columns held by each 16-lane half
    #pragma unroll
    for (int r = 0; r < 8; ++r) {
        float x = partial[r];
        x += __shfl_xor(x, 1, 32);
        x += __shfl_xor(x, 2, 32);
        x += __shfl_xor(x, 4, 32);
        x += __shfl_xor(x, 8, 32);
        partial[r] = x;
    }
    if (col == 0) {   // lanes 0 (rows 0..7) and 16 (rows 8..15) of each wave
        #pragma unroll
        for (int r = 0; r < 8; ++r) {
            const int row = mbase + wave * 16 + half * 8 + r;
            logits[row] = partial[r] + bVv;
        }
    }
}

// ---------------- Kernel 4: softmax over L per batch ----------------
__global__ void __launch_bounds__(256) softmax_kernel(const float* __restrict__ logits,
                                                      float* __restrict__ attn) {
    __shared__ float red[8];
    const int b = blockIdx.x, tid = threadIdx.x;
    const int wave = tid >> 5, lane = tid & 31;
    float v[4];
    float mx = -3.402823466e38f;
    #pragma unroll
    for (int r = 0; r < 4; ++r) {
        v[r] = logits[(size_t)b * Lc + r * 256 + tid];
        mx = fmaxf(mx, v[r]);
    }
    #pragma unroll
    for (int off = 16; off >= 1; off >>= 1) mx = fmaxf(mx, __shfl_xor(mx, off, 32));
    if (lane == 0) red[wave] = mx;
    __syncthreads();
    float bm = red[0];
    #pragma unroll
    for (int w = 1; w < 8; ++w) bm = fmaxf(bm, red[w]);
    float s = 0.f;
    #pragma unroll
    for (int r = 0; r < 4; ++r) { v[r] = __expf(v[r] - bm); s += v[r]; }
    #pragma unroll
    for (int off = 16; off >= 1; off >>= 1) s += __shfl_xor(s, off, 32);
    __syncthreads();
    if (lane == 0) red[wave] = s;
    __syncthreads();
    float bs = 0.f;
    #pragma unroll
    for (int w = 0; w < 8; ++w) bs += red[w];
    const float inv = 1.0f / bs;
    #pragma unroll
    for (int r = 0; r < 4; ++r)
        attn[(size_t)b * Lc + r * 256 + tid] = v[r] * inv;
}

// ---------------- Kernel 5: context[b][d] = sum_l attn * features ----------------
__global__ void __launch_bounds__(256) context_kernel(const float* __restrict__ features,
                                                      const float* __restrict__ attn,
                                                      float* __restrict__ ctx) {
    __shared__ float sa[Lc];
    const int b = blockIdx.y;
    const int d = blockIdx.x * 256 + threadIdx.x;
    for (int i = threadIdx.x; i < Lc; i += 256) sa[i] = attn[(size_t)b * Lc + i];
    __syncthreads();
    const float* f = features + (size_t)b * Lc * Dc + d;
    float acc = 0.f;
    #pragma unroll 4
    for (int l = 0; l < Lc; ++l)
        acc += sa[l] * f[(size_t)l * Dc];
    ctx[(size_t)b * Dc + d] = acc;
}

extern "C" void kernel_launch(void* const* d_in, const int* in_sizes, int n_in,
                              void* d_out, int out_size, void* d_ws, size_t ws_size,
                              hipStream_t stream) {
    const float* features = (const float*)d_in[0];
    const float* hidden   = (const float*)d_in[1];
    const float* W1       = (const float*)d_in[2];
    const float* b1       = (const float*)d_in[3];
    const float* W2       = (const float*)d_in[4];
    const float* b2       = (const float*)d_in[5];
    const float* V        = (const float*)d_in[6];
    const float* bV       = (const float*)d_in[7];

    // workspace layout
    unsigned short* W1T = (unsigned short*)d_ws;                               // 2 MB
    float* ph     = (float*)((char*)d_ws + (size_t)Uc * Dc * 2);               // 256 KB
    float* logits = (float*)((char*)ph + (size_t)Bc * Uc * 4);                 // 256 KB

    float* ctx  = (float*)d_out;            // [B, D]
    float* attn = (float*)d_out + Bc * Dc;  // [B, L, 1]

    transpose_w1_kernel<<<dim3(Uc / 32, Dc / 32), dim3(32, 8), 0, stream>>>(W1, W1T);
    proj_h_kernel<<<dim3(Uc / 256, Bc), 256, 0, stream>>>(hidden, W2, b2, b1, ph);

    (void)hipFuncSetAttribute(reinterpret_cast<const void*>(attn_logits_wmma),
                              hipFuncAttributeMaxDynamicSharedMemorySize, LDS_BYTES);
    attn_logits_wmma<<<ROWS_TOTAL / MTILE, NTHREADS, LDS_BYTES, stream>>>(features, W1T, ph, V, bV, logits);

    softmax_kernel<<<Bc, 256, 0, stream>>>(logits, attn);
    context_kernel<<<dim3(Dc / 256, Bc), 256, 0, stream>>>(features, attn, ctx);
}
